// TernaryLinear_23682449670371
// MI455X (gfx1250) — matrix operand order repaired and checked by reading, not compiled, not run on hardware
//
#include <hip/hip_runtime.h>
#include <math.h>

// MI455X / gfx1250, wave32, WMMA f16 path, double-buffered LDS pipeline.
// y[b,o] = scale[o] * sum_i x[b,i]*tern[o,i] + bias[o]
// tern in {-1,0,+1} exactly representable in f16 -> only x is rounded to f16.

typedef __attribute__((ext_vector_type(16))) _Float16 v16h;
typedef __attribute__((ext_vector_type(8)))  _Float16 v8h;
typedef __attribute__((ext_vector_type(4)))  _Float16 v4h;
typedef __attribute__((ext_vector_type(8)))  float    v8f;
typedef __attribute__((ext_vector_type(4)))  float    v4f;
typedef __attribute__((ext_vector_type(4)))  int      v4i;

constexpr int K_IN   = 4096;  // inner dim
constexpr int N_OUT  = 1024;  // output features
constexpr int M_B    = 4096;  // batch
constexpr float THRESH = 0.05f;

#if __has_builtin(__builtin_amdgcn_global_load_async_to_lds_b128)
#define TERN_ASYNC_LDS 1
#endif

__device__ __forceinline__ void wait_async_lds() {
#if defined(TERN_ASYNC_LDS)
#if __has_builtin(__builtin_amdgcn_s_wait_asynccnt)
  __builtin_amdgcn_s_wait_asynccnt(0);
#else
  asm volatile("s_wait_asynccnt 0" ::: "memory");
#endif
#endif
}

// ---------------------------------------------------------------------------
// Kernel 1: per-row ternary quantization of W [N_OUT, K_IN] (f32)
//   tern[o,i] in {-1,0,+1} as f16,  scale[o] = mean |w| over kept entries
// ---------------------------------------------------------------------------
__global__ __launch_bounds__(256)
void tern_quant_kernel(const float* __restrict__ w,
                       _Float16* __restrict__ tern,
                       float* __restrict__ scale) {
  __shared__ float red[256];
  __shared__ float s_delta;
  const int o = blockIdx.x;
  const int t = threadIdx.x;
  const float* row = w + (size_t)o * K_IN;

  float s = 0.f;
  for (int i = t * 4; i < K_IN; i += 256 * 4) {
    v4f v = *(const v4f*)(row + i);
    s += fabsf(v.x) + fabsf(v.y) + fabsf(v.z) + fabsf(v.w);
  }
  red[t] = s;
  __syncthreads();
  for (int off = 128; off > 0; off >>= 1) {
    if (t < off) red[t] += red[t + off];
    __syncthreads();
  }
  if (t == 0) s_delta = THRESH * red[0] / (float)K_IN;
  __syncthreads();
  const float delta = s_delta;

  float kept = 0.f, cnt = 0.f;
  for (int i = t * 4; i < K_IN; i += 256 * 4) {
    v4f v = *(const v4f*)(row + i);
    v4h q;
#pragma unroll
    for (int j = 0; j < 4; j++) {
      float xv = v[j];
      float a  = fabsf(xv);
      bool keep = a > delta;
      q[j] = (_Float16)(keep ? (xv > 0.f ? 1.f : -1.f) : 0.f);
      kept += keep ? a : 0.f;
      cnt  += keep ? 1.f : 0.f;
    }
    *(v4h*)(tern + (size_t)o * K_IN + i) = q;
  }
  red[t] = kept;
  __syncthreads();
  for (int off = 128; off > 0; off >>= 1) {
    if (t < off) red[t] += red[t + off];
    __syncthreads();
  }
  const float keptsum = red[0];
  __syncthreads();
  red[t] = cnt;
  __syncthreads();
  for (int off = 128; off > 0; off >>= 1) {
    if (t < off) red[t] += red[t + off];
    __syncthreads();
  }
  if (t == 0) scale[o] = keptsum / fmaxf(red[0], 1.f);
}

// ---------------------------------------------------------------------------
// Kernel 2: WMMA GEMM, block tile 128x128, 8 waves (2M x 4N), wave tile 64x32.
// Double-buffered LDS; B tile copied global->LDS with async-to-LDS (ASYNCcnt),
// A tile staged through VGPRs for the f32->f16 convert.
// ---------------------------------------------------------------------------
constexpr int BM = 128;
constexpr int BN = 128;
constexpr int BK = 32;
constexpr int A_STR = 40;  // halves per LDS row (32 + 8 pad): conflict-free frag reads
constexpr int B_STR = 40;
constexpr int NK = K_IN / BK;  // 128 (even)

__device__ __forceinline__ void load_a_regs(const float* __restrict__ x,
                                            int m0, int k0, int t, v4f areg[4]) {
#pragma unroll
  for (int i = 0; i < 4; i++) {
    int f = t + i * 256;  // float4 id: 128 rows * 8 quads
    int r = f >> 3;
    int c = (f & 7) << 2;
    areg[i] = *(const v4f*)(x + (size_t)(m0 + r) * K_IN + k0 + c);
  }
}

__device__ __forceinline__ void store_a_lds(_Float16* __restrict__ lA,
                                            int t, const v4f areg[4]) {
#pragma unroll
  for (int i = 0; i < 4; i++) {
    int f = t + i * 256;
    int r = f >> 3;
    int c = (f & 7) << 2;
    v4h h;
    h[0] = (_Float16)areg[i][0]; h[1] = (_Float16)areg[i][1];
    h[2] = (_Float16)areg[i][2]; h[3] = (_Float16)areg[i][3];
    *(v4h*)(&lA[r * A_STR + c]) = h;  // ds_store_b64 (v_cvt_pk_f16_f32 pairs)
  }
}

// B tile: 128 rows x 32 halves = 512 x 16B chunks; 2 chunks per thread.
__device__ __forceinline__ void stage_b(const _Float16* __restrict__ tern,
                                        _Float16* __restrict__ lB,
                                        int n0, int k0, int t) {
#pragma unroll
  for (int i = 0; i < 2; i++) {
    int f = t + i * 256;
    int r = f >> 2;
    int c = (f & 3) << 3;
    const _Float16* g = tern + (size_t)(n0 + r) * K_IN + k0 + c;
    _Float16* l = &lB[r * B_STR + c];
#if defined(TERN_ASYNC_LDS)
    // direct global->LDS DMA, no VGPR data movement (tracked by ASYNCcnt)
    __builtin_amdgcn_global_load_async_to_lds_b128(
        (v4i*)(void*)g, (v4i*)(void*)l, 0 /*imm offset*/, 0 /*cpol*/);
#else
    *(v8h*)l = *(const v8h*)g;
#endif
  }
}

__device__ __forceinline__ void compute_tile(const _Float16* __restrict__ lA,
                                             const _Float16* __restrict__ lB,
                                             int wm, int wn, int l16, int lhalf,
                                             v8f acc[4][2]) {
  // A 16x32 frag: lane row m=l16; halves 0..7 = K[8*lhalf..+7], 8..15 = K[16+8*lhalf..+7]
  v16h afrag[4];
#pragma unroll
  for (int mi = 0; mi < 4; mi++) {
    int m   = wm * 64 + mi * 16 + l16;
    int klo = lhalf * 8;
    v8h lo = *(const v8h*)(&lA[m * A_STR + klo]);
    v8h hi = *(const v8h*)(&lA[m * A_STR + 16 + klo]);
    v16h a;
#pragma unroll
    for (int j = 0; j < 8; j++) { a[j] = lo[j]; a[j + 8] = hi[j]; }
    afrag[mi] = a;
  }
  // B 32x16 frag: lane col n=l16; halves 0..7 = K[8*lhalf..+7], 8..15 = K[16+8*lhalf..+7]
  v16h bfrag[2];
#pragma unroll
  for (int ni = 0; ni < 2; ni++) {
    int n  = wn * 32 + ni * 16 + l16;
    int kb = lhalf * 8;
    v8h lo = *(const v8h*)(&lB[n * B_STR + kb]);
    v8h hi = *(const v8h*)(&lB[n * B_STR + kb + 16]);
    v16h b;
#pragma unroll
    for (int j = 0; j < 8; j++) { b[j] = lo[j]; b[j + 8] = hi[j]; }
    bfrag[ni] = b;
  }
#pragma unroll
  for (int mi = 0; mi < 4; mi++)
#pragma unroll
    for (int ni = 0; ni < 2; ni++)
      acc[mi][ni] = __builtin_amdgcn_wmma_f32_16x16x32_f16(
          false, afrag[mi], false, bfrag[ni],
          (short)0, acc[mi][ni], false, false);
}

__global__ __launch_bounds__(256)
void tern_gemm_kernel(const float* __restrict__ x,
                      const _Float16* __restrict__ tern,
                      const float* __restrict__ scale,
                      const float* __restrict__ bias,
                      float* __restrict__ out) {
  __shared__ _Float16 lA[2][BM * A_STR];  // 2 x 10 KB
  __shared__ _Float16 lB[2][BN * B_STR];  // 2 x 10 KB

  const int t     = threadIdx.x;
  const int lane  = t & 31;
  const int wave  = t >> 5;
  const int wm    = wave & 1;   // 0..1 -> 64 M rows each
  const int wn    = wave >> 1;  // 0..3 -> 32 N cols each
  const int l16   = lane & 15;
  const int lhalf = lane >> 4;
  const int m0    = blockIdx.y * BM;
  const int n0    = blockIdx.x * BN;

  v8f acc[4][2];
#pragma unroll
  for (int i = 0; i < 4; i++)
#pragma unroll
    for (int j = 0; j < 2; j++) acc[i][j] = {};

  v4f areg[4];

  // ---- prologue: stage tile 0 into buffer 0
  load_a_regs(x, m0, 0, t, areg);
  stage_b(tern, lB[0], n0, 0, t);
  store_a_lds(lA[0], t, areg);
  wait_async_lds();
  __syncthreads();

  // ---- main pipeline, unrolled x2 so LDS buffer addresses are immediates
  for (int k = 0; k < NK; k += 2) {
    // compute buf0 while staging tile k+1 into buf1 (k+1 < NK always: NK even)
    load_a_regs(x, m0, (k + 1) * BK, t, areg);
    stage_b(tern, lB[1], n0, (k + 1) * BK, t);
    if (k + 2 < NK)  // L2 prefetch two tiles ahead for A
      __builtin_prefetch(x + (size_t)(m0 + (t >> 3)) * K_IN + (k + 2) * BK + ((t & 7) << 2), 0, 3);
    compute_tile(lA[0], lB[0], wm, wn, l16, lhalf, acc);
    store_a_lds(lA[1], t, areg);
    wait_async_lds();
    __syncthreads();

    // compute buf1 while staging tile k+2 into buf0
    const bool more = (k + 2 < NK);
    if (more) {
      load_a_regs(x, m0, (k + 2) * BK, t, areg);
      stage_b(tern, lB[0], n0, (k + 2) * BK, t);
    }
    compute_tile(lA[1], lB[1], wm, wn, l16, lhalf, acc);
    if (more) store_a_lds(lA[0], t, areg);
    wait_async_lds();
    __syncthreads();
  }

  // ---- epilogue: y = acc * scale[col] + bias[col]
  // C/D layout: vgpr r of lane -> M = r + 8*lhalf, N = l16
#pragma unroll
  for (int ni = 0; ni < 2; ni++) {
    int col  = n0 + wn * 32 + ni * 16 + l16;
    float sc = scale[col];
    float bs = bias[col];
#pragma unroll
    for (int mi = 0; mi < 4; mi++) {
      int rbase = m0 + wm * 64 + mi * 16 + lhalf * 8;
#pragma unroll
      for (int r = 0; r < 8; r++) {
        out[(size_t)(rbase + r) * N_OUT + col] = acc[mi][ni][r] * sc + bs;
      }
    }
  }
}

// ---------------------------------------------------------------------------
// Launch: ws layout = [ scale: 1024 f32 (4KB) | tern: 1024*4096 f16 (8MB) ]
// ---------------------------------------------------------------------------
extern "C" void kernel_launch(void* const* d_in, const int* in_sizes, int n_in,
                              void* d_out, int out_size, void* d_ws, size_t ws_size,
                              hipStream_t stream) {
  (void)in_sizes; (void)n_in; (void)out_size; (void)ws_size;
  const float* x    = (const float*)d_in[0];
  const float* w    = (const float*)d_in[1];
  const float* bias = (const float*)d_in[2];
  float* out        = (float*)d_out;

  float*    scale = (float*)d_ws;
  _Float16* tern  = (_Float16*)((char*)d_ws + N_OUT * sizeof(float));

  tern_quant_kernel<<<N_OUT, 256, 0, stream>>>(w, tern, scale);

  dim3 grid(N_OUT / BN, M_B / BM);  // (8, 32)
  tern_gemm_kernel<<<grid, 256, 0, stream>>>(x, tern, scale, bias, out);
}
